// ChannelWiseMamba_31009663877239
// MI455X (gfx1250) — matrix operand order repaired and checked by reading, not compiled, not run on hardware
//
#include <hip/hip_runtime.h>
#include <hip/hip_bf16.h>

// ---------------------------------------------------------------------------
// CDNA5 (gfx1250) Mamba block.
//   M = BATCH*CH*SEQ = 32768 token rows.
//   GEMMs use v_wmma_f32_16x16x32_bf16 (bf16 in, fp32 accumulate).
//   Fragments are loaded straight from global memory with b128 loads whose
//   lane mapping matches the ISA A/B VGPR layouts (weights are L2-resident:
//   192MB L2 >> ~3.3MB of bf16 weights).
// ---------------------------------------------------------------------------

typedef __attribute__((ext_vector_type(16))) __bf16 v16bf;
typedef __attribute__((ext_vector_type(8)))  float  v8f;

union Frag {
  uint4 q[2];   // 8 VGPRs = 16 bf16 per lane
  v16bf v;
};

__device__ __forceinline__ unsigned short f2bf(float f) {
  unsigned u = __float_as_uint(f);
  u += 0x7FFFu + ((u >> 16) & 1u);          // round-to-nearest-even
  return (unsigned short)(u >> 16);
}
__device__ __forceinline__ float bf2f(unsigned short h) {
  return __uint_as_float(((unsigned)h) << 16);
}
__device__ __forceinline__ uint4 pack8f(float4 a, float4 b) {
  uint4 r;
  r.x = (unsigned)f2bf(a.x) | ((unsigned)f2bf(a.y) << 16);
  r.y = (unsigned)f2bf(a.z) | ((unsigned)f2bf(a.w) << 16);
  r.z = (unsigned)f2bf(b.x) | ((unsigned)f2bf(b.y) << 16);
  r.w = (unsigned)f2bf(b.z) | ((unsigned)f2bf(b.w) << 16);
  return r;
}

// ---------------------------------------------------------------------------
__global__ __launch_bounds__(256)
void k_cvt_f32_bf16(const float* __restrict__ in, unsigned short* __restrict__ out, int n) {
  for (int i = blockIdx.x * blockDim.x + threadIdx.x; i < n; i += gridDim.x * blockDim.x)
    out[i] = f2bf(in[i]);
}

// ---------------------------------------------------------------------------
// Generic WMMA GEMM:  C[M,N] = A[M,K] * B[N,K]^T
//   BM x BN block tile, 8 waves (wave32) each owning a 64(M) x 32(N) tile
//   = 4x2 sub-tiles of 16x16, K stepped by 32.
// A fragment (16x32 bf16): lane<16 -> M=lane, K={k0..k0+7, k0+16..k0+23};
//                          lane>=16 -> M=lane-16, K={k0+8..15, k0+24..31}.
// B fragment (32x16 bf16): lane -> N=lane&15, K same as A fragment.
// Both map to two contiguous 16-byte global loads per fragment.
// EPI: 0 = store bf16, 1 = store fp32, 2 = softplus(acc + bias[n]) fp32.
// ---------------------------------------------------------------------------
template<int BM, int BN, bool A_F32, int EPI>
__global__ __launch_bounds__(256)
void k_gemm(const void* __restrict__ Aptr, const unsigned short* __restrict__ Bw,
            void* __restrict__ Cptr, int M, int N, int K, int lda,
            const float* __restrict__ bias)
{
  constexpr int WCOLS = BN / 32;
  constexpr int WROWS = BM / 64;
  static_assert(WROWS * WCOLS == 8, "8 waves per block");

  const int lane  = threadIdx.x & 31;
  const int wid   = threadIdx.x >> 5;
  const int wm    = wid / WCOLS;
  const int wn    = wid % WCOLS;
  const int blkM  = blockIdx.y * BM;
  const int blkN  = blockIdx.x * BN;
  const int mHalf = lane >> 4;
  const int l16   = lane & 15;
  const int aKh   = mHalf * 8;
  const int bKh   = mHalf * 8;

  long aRow[4];
#pragma unroll
  for (int i = 0; i < 4; ++i)
    aRow[i] = (long)(blkM + wm * 64 + i * 16 + l16) * lda;
  long bCol[2];
#pragma unroll
  for (int j = 0; j < 2; ++j)
    bCol[j] = (long)(blkN + wn * 32 + j * 16 + l16) * K;   // B row stride == K

  const unsigned short* Ab = (const unsigned short*)Aptr;
  const float*          Af = (const float*)Aptr;

  v8f acc[4][2] = {};

  for (int k0 = 0; k0 < K; k0 += 32) {
    Frag a[4], b[2];
#pragma unroll
    for (int i = 0; i < 4; ++i) {
      if constexpr (A_F32) {
        const float* p0 = Af + aRow[i] + k0 + aKh;
        float4 f0 = *(const float4*)(p0);
        float4 f1 = *(const float4*)(p0 + 4);
        float4 f2 = *(const float4*)(p0 + 16);
        float4 f3 = *(const float4*)(p0 + 20);
        a[i].q[0] = pack8f(f0, f1);
        a[i].q[1] = pack8f(f2, f3);
      } else {
        const unsigned short* p0 = Ab + aRow[i] + k0 + aKh;
        a[i].q[0] = *(const uint4*)(p0);
        a[i].q[1] = *(const uint4*)(p0 + 16);
      }
    }
#pragma unroll
    for (int j = 0; j < 2; ++j) {
      const unsigned short* p0 = Bw + bCol[j] + k0 + bKh;
      b[j].q[0] = *(const uint4*)(p0);
      b[j].q[1] = *(const uint4*)(p0 + 16);
    }
    if (k0 + 32 < K) {  // stream next A tile through L2 (global_prefetch_b8)
      if constexpr (A_F32) __builtin_prefetch(Af + aRow[0] + k0 + 32 + aKh, 0, 3);
      else                 __builtin_prefetch(Ab + aRow[0] + k0 + 32 + aKh, 0, 3);
    }
#pragma unroll
    for (int i = 0; i < 4; ++i)
#pragma unroll
      for (int j = 0; j < 2; ++j)
        acc[i][j] = __builtin_amdgcn_wmma_f32_16x16x32_bf16(
            false, a[i].v, false, b[j].v, (short)0, acc[i][j], false, false);
  }

  // D layout: VGPR r -> M = r + (lane>>4)*8 ; N = lane&15
#pragma unroll
  for (int i = 0; i < 4; ++i) {
    const int rBase = blkM + wm * 64 + i * 16 + mHalf * 8;
#pragma unroll
    for (int j = 0; j < 2; ++j) {
      const int col = blkN + wn * 32 + j * 16 + l16;
#pragma unroll
      for (int r = 0; r < 8; ++r) {
        const long idx = (long)(rBase + r) * N + col;
        const float v = acc[i][j][r];
        if constexpr (EPI == 0) {
          ((unsigned short*)Cptr)[idx] = f2bf(v);
        } else if constexpr (EPI == 1) {
          ((float*)Cptr)[idx] = v;
        } else {
          const float x = v + bias[col];
          ((float*)Cptr)[idx] = (x > 20.f) ? x : log1pf(__expf(x));
        }
      }
    }
  }
}

// ---------------------------------------------------------------------------
// Depthwise causal conv (D_CONV=4) + SiLU. One thread per (seq, channel),
// marching t with a register shift window; lanes cover 32 consecutive
// channels -> coalesced 64B loads / 128B stores per wave per step.
// ---------------------------------------------------------------------------
__global__ __launch_bounds__(256)
void k_conv_silu(const unsigned short* __restrict__ xz, const float* __restrict__ cw,
                 const float* __restrict__ cbias, float* __restrict__ xc)
{
  const int gid = blockIdx.x * blockDim.x + threadIdx.x;  // 32768 = 32 seq * 1024 ch
  const int seq = gid >> 10;
  const int d   = gid & 1023;
  const float w0 = cw[d * 4 + 0], w1 = cw[d * 4 + 1];
  const float w2 = cw[d * 4 + 2], w3 = cw[d * 4 + 3];
  const float bias = cbias[d];
  float x0 = 0.f, x1 = 0.f, x2 = 0.f;
  const unsigned short* col = xz + (long)seq * 1024 * 2048 + d;  // xs = xz[:, :1024]
  float* out = xc + (long)seq * 1024 * 1024 + d;
  for (int t = 0; t < 1024; ++t) {
    const float x3 = bf2f(col[(long)t * 2048]);
    const float v = w0 * x0 + w1 * x1 + w2 * x2 + w3 * x3 + bias;
    out[(long)t * 1024] = v / (1.f + __expf(-v));   // SiLU
    x0 = x1; x1 = x2; x2 = x3;
  }
}

// ---------------------------------------------------------------------------
// Selective scan. One thread per (seq, channel): h[16] in VGPRs, L=1024 steps.
// Per step, each lane loads one of the 32 shared B/C floats (one coalesced
// 128B load per wave) and broadcasts with wave shuffles. Epilogue fuses the
// +xs*D residual and the silu(z) gate, stores bf16 for the out-proj GEMM.
// ---------------------------------------------------------------------------
__global__ __launch_bounds__(256)
void k_scan(const float* __restrict__ dt, const float* __restrict__ xc,
            const float* __restrict__ xdbl, const unsigned short* __restrict__ xz,
            const float* __restrict__ A_log, const float* __restrict__ Dvec,
            unsigned short* __restrict__ yg)
{
  const int gid  = blockIdx.x * blockDim.x + threadIdx.x;  // 32768 lanes
  const int seq  = gid >> 10;
  const int d    = gid & 1023;
  const int lane = threadIdx.x & 31;

  float An[16];
#pragma unroll
  for (int s = 0; s < 16; ++s) An[s] = -__expf(A_log[d * 16 + s]);  // A = -exp(A_log)
  const float Dd = Dvec[d];

  float h[16];
#pragma unroll
  for (int s = 0; s < 16; ++s) h[s] = 0.f;

  const long mBase = (long)seq * 1024;
  for (int t = 0; t < 1024; ++t) {
    const long m  = mBase + t;
    const float dtv = dt[m * 1024 + d];
    const float xv  = xc[m * 1024 + d];
    const float bc  = xdbl[m * 64 + 32 + lane];   // lanes 0-15: B, 16-31: C
    const float dtx = dtv * xv;
    float y = 0.f;
#pragma unroll
    for (int s = 0; s < 16; ++s) {
      const float Bs = __shfl(bc, s, 32);
      const float Cs = __shfl(bc, s + 16, 32);
      h[s] = __expf(dtv * An[s]) * h[s] + dtx * Bs;
      y = fmaf(h[s], Cs, y);
    }
    const float zf  = bf2f(xz[m * 2048 + 1024 + d]);     // z = xz[:, 1024:]
    const float gate = zf / (1.f + __expf(-zf));          // silu(z)
    yg[m * 1024 + d] = f2bf((y + xv * Dd) * gate);
  }
}

// ---------------------------------------------------------------------------
extern "C" void kernel_launch(void* const* d_in, const int* in_sizes, int n_in,
                              void* d_out, int out_size, void* d_ws, size_t ws_size,
                              hipStream_t stream)
{
  const float* x      = (const float*)d_in[0];  // (4,8,1024,512)
  const float* W_in   = (const float*)d_in[1];  // (2048,512)
  const float* conv_w = (const float*)d_in[2];  // (1024,4)
  const float* conv_b = (const float*)d_in[3];  // (1024,)
  const float* W_xp   = (const float*)d_in[4];  // (64,1024)
  const float* W_dt   = (const float*)d_in[5];  // (1024,32)
  const float* b_dt   = (const float*)d_in[6];  // (1024,)
  const float* A_log  = (const float*)d_in[7];  // (1024,16)
  const float* Dvec   = (const float*)d_in[8];  // (1024,)
  const float* W_out  = (const float*)d_in[9];  // (512,1024)
  float* out = (float*)d_out;                   // (4,8,1024,512) fp32

  constexpr long M    = 32768;  // 4*8*1024 token rows
  constexpr int  DM   = 512, DI = 1024, NXZ = 2048, NDBL = 64, DTR = 32;

  // workspace carve-out (~520 MB)
  char* ws = (char*)d_ws;
  auto carve = [&](size_t bytes) -> char* {
    char* p = ws; ws += (bytes + 255) & ~(size_t)255; return p;
  };
  unsigned short* x_bf    = (unsigned short*)carve((size_t)M * DM * 2);
  unsigned short* Win_bf  = (unsigned short*)carve((size_t)NXZ * DM * 2);
  unsigned short* Wxp_bf  = (unsigned short*)carve((size_t)NDBL * DI * 2);
  unsigned short* Wdt_bf  = (unsigned short*)carve((size_t)DI * DTR * 2);
  unsigned short* Wout_bf = (unsigned short*)carve((size_t)DM * DI * 2);
  unsigned short* xz_bf   = (unsigned short*)carve((size_t)M * NXZ * 2);
  float*          xc      = (float*)carve((size_t)M * DI * 4);
  float*          xdbl    = (float*)carve((size_t)M * NDBL * 4);
  float*          dt      = (float*)carve((size_t)M * DI * 4);
  unsigned short* yg_bf   = (unsigned short*)carve((size_t)M * DI * 2);
  (void)in_sizes; (void)n_in; (void)out_size; (void)ws_size;

  // fp32 -> bf16 converts (weights then stay hot in 192MB L2)
  k_cvt_f32_bf16<<<2048, 256, 0, stream>>>(x,     x_bf,    (int)(M * DM));
  k_cvt_f32_bf16<<<256,  256, 0, stream>>>(W_in,  Win_bf,  NXZ * DM);
  k_cvt_f32_bf16<<<64,   256, 0, stream>>>(W_xp,  Wxp_bf,  NDBL * DI);
  k_cvt_f32_bf16<<<32,   256, 0, stream>>>(W_dt,  Wdt_bf,  DI * DTR);
  k_cvt_f32_bf16<<<128,  256, 0, stream>>>(W_out, Wout_bf, DM * DI);

  const dim3 blk(256);
  // 1) in-proj: xz[M,2048] = x[M,512] @ W_in^T        (bf16 out)
  k_gemm<128, 128, false, 0><<<dim3(NXZ / 128, M / 128), blk, 0, stream>>>(
      x_bf, Win_bf, xz_bf, (int)M, NXZ, DM, DM, nullptr);
  // 2) depthwise conv + SiLU -> xc fp32
  k_conv_silu<<<128, 256, 0, stream>>>(xz_bf, conv_w, conv_b, xc);
  // 3) x-proj: xdbl[M,64] = xc @ W_xproj^T            (fp32 A cvt on the fly)
  k_gemm<256, 64, true, 1><<<dim3(NDBL / 64, M / 256), blk, 0, stream>>>(
      xc, Wxp_bf, xdbl, (int)M, NDBL, DI, DI, nullptr);
  // 4) dt-proj: dt[M,1024] = softplus(xdbl[:, :32] @ W_dt^T + b_dt)
  k_gemm<128, 128, true, 2><<<dim3(DI / 128, M / 128), blk, 0, stream>>>(
      xdbl, Wdt_bf, dt, (int)M, DI, DTR, NDBL, b_dt);
  // 5) selective scan + residual + gate -> yg bf16
  k_scan<<<128, 256, 0, stream>>>(dt, xc, xdbl, xz_bf, A_log, Dvec, yg_bf);
  // 6) out-proj: out[M,512] = yg @ W_out^T            (fp32 out)
  k_gemm<128, 128, false, 1><<<dim3(DM / 128, M / 128), blk, 0, stream>>>(
      yg_bf, Wout_bf, out, (int)M, DM, DI, DI, nullptr);
}
